// GATLayer_57475252355080
// MI455X (gfx1250) — matrix operand order repaired and checked by reading, not compiled, not run on hardware
//
#include <hip/hip_runtime.h>

typedef __attribute__((ext_vector_type(16))) _Float16 v16h;
typedef __attribute__((ext_vector_type(8)))  _Float16 v8h;
typedef __attribute__((ext_vector_type(8)))  float    v8f;
typedef int v4i __attribute__((vector_size(16)));   // matches builtin param type

#define N_NODES 4096
#define F_IN    256
#define F_OUT   64
#define H_HEADS 4
#define NEG_INF (-1000000000.0f)

#define AS1 __attribute__((address_space(1)))
#define AS3 __attribute__((address_space(3)))

#if __has_builtin(__builtin_amdgcn_global_load_async_to_lds_b128) && \
    __has_builtin(__builtin_amdgcn_s_wait_asynccnt)
#define GAT_ASYNC 1
#else
#define GAT_ASYNC 0
#endif

__device__ __forceinline__ float leaky(float e) { return e >= 0.0f ? e : 0.2f * e; }

// ---------------------------------------------------------------------------
// Kernel 0: WT[h][o][k] = (f16) W[h][k][o]   (contiguous-k rows for WMMA B)
// ---------------------------------------------------------------------------
__global__ __launch_bounds__(256) void prep_wt(const float* __restrict__ W,
                                               _Float16* __restrict__ WT) {
    int tid = blockIdx.x * blockDim.x + threadIdx.x;   // H*F_IN*F_OUT = 65536
    int h   = tid >> 14;                               // 16384 per head
    int rem = tid & 16383;
    int o   = rem >> 8;                                // 0..63
    int k   = rem & 255;                               // 0..255
    WT[(h * F_OUT + o) * F_IN + k] = (_Float16)W[(h * F_IN + k) * F_OUT + o];
}

// ---------------------------------------------------------------------------
// Kernel 1: hT[h][o][n] = (f16) (x @ W[h])[n][o]  via v_wmma_f32_16x16x32_f16
// One wave = one 16x16 output tile, K-loop over F_IN in steps of 32.
// ---------------------------------------------------------------------------
__global__ __launch_bounds__(256) void gemm_h(const float* __restrict__ x,
                                              const _Float16* __restrict__ WT,
                                              _Float16* __restrict__ hT) {
    int wave = (blockIdx.x * blockDim.x + threadIdx.x) >> 5;  // 4096 waves
    int lane = threadIdx.x & 31;
    int head = wave >> 10;            // 1024 waves per head
    int rem  = wave & 1023;
    int rowTile = rem >> 2;           // 0..255 (16-row tiles of n)
    int colTile = rem & 3;            // 0..3   (16-col tiles of o)
    int m  = lane & 15;
    int kh = lane >> 4;               // lane group selects K-halves (A layout)
    int row = rowTile * 16 + m;
    int col = colTile * 16 + m;       // B/C column for this lane (N = lane%16)

    const float*    xrow = x  + (size_t)row * F_IN;
    const _Float16* wrow = WT + ((size_t)head * F_OUT + col) * F_IN;

    v8f acc = {};
    for (int kb = 0; kb < F_IN; kb += 32) {
        // A: 16x32 f16 tile of x. Lane(kh) holds K = 8kh..+7 (elems 0..7) and
        // K = 16+8kh..+7 (elems 8..15) per ISA A-layout.
        float af[16];
        *(float4*)(af + 0)  = *(const float4*)(xrow + kb + 8 * kh);
        *(float4*)(af + 4)  = *(const float4*)(xrow + kb + 8 * kh + 4);
        *(float4*)(af + 8)  = *(const float4*)(xrow + kb + 16 + 8 * kh);
        *(float4*)(af + 12) = *(const float4*)(xrow + kb + 16 + 8 * kh + 4);
        v16h A;
#pragma unroll
        for (int e = 0; e < 16; ++e) A[e] = (_Float16)af[e];

        // B: 32x16 f16 tile of W. Lane holds K = 8kh..+7 (elems 0..7), 16+8kh..+7 (elems 8..15).
        v8h b0 = *(const v8h*)(wrow + kb + 8 * kh);
        v8h b1 = *(const v8h*)(wrow + kb + 16 + 8 * kh);
        v16h B;
#pragma unroll
        for (int e = 0; e < 8; ++e) { B[e] = b0[e]; B[8 + e] = b1[e]; }

        acc = __builtin_amdgcn_wmma_f32_16x16x32_f16(false, A, false, B,
                                                     (short)0, acc, false, false);
    }

    // C layout: VGPR r, lane group kh -> M = r + 8*kh, N = lane%16
#pragma unroll
    for (int r = 0; r < 8; ++r) {
        int n = rowTile * 16 + r + 8 * kh;
        hT[((size_t)head * F_OUT + col) * N_NODES + n] = (_Float16)acc[r];
    }
}

// ---------------------------------------------------------------------------
// Kernel 2: f1[h][n] = h[h][n][:].a1[h], f2 likewise with a2
// ---------------------------------------------------------------------------
__global__ __launch_bounds__(256) void f1f2_k(const _Float16* __restrict__ hT,
                                              const float* __restrict__ a,
                                              float* __restrict__ f1,
                                              float* __restrict__ f2) {
    int tid = blockIdx.x * blockDim.x + threadIdx.x;   // H*N = 16384
    int h = tid >> 12;
    int n = tid & (N_NODES - 1);
    float s1 = 0.0f, s2 = 0.0f;
#pragma unroll 8
    for (int o = 0; o < F_OUT; ++o) {
        float hv = (float)hT[((size_t)h * F_OUT + o) * N_NODES + n];
        s1 += hv * a[h * 2 * F_OUT + o];
        s2 += hv * a[h * 2 * F_OUT + F_OUT + o];
    }
    f1[tid] = s1;
    f2[tid] = s2;
}

// ---------------------------------------------------------------------------
// Kernel 3: streaming per-row softmax stats (flash pass 1).
// ---------------------------------------------------------------------------
__global__ __launch_bounds__(256) void row_stats(const int* __restrict__ adj,
                                                 const float* __restrict__ f1,
                                                 const float* __restrict__ f2,
                                                 float* __restrict__ rowM,
                                                 float* __restrict__ rowIS) {
    int i = blockIdx.x;
    int t = threadIdx.x;
    __shared__ float shm[H_HEADS][256];
    __shared__ float shs[H_HEADS][256];

    float f1i[H_HEADS], m[H_HEADS], s[H_HEADS];
#pragma unroll
    for (int h = 0; h < H_HEADS; ++h) {
        f1i[h] = f1[h * N_NODES + i];
        m[h] = -3.0e38f;
        s[h] = 0.0f;
    }

    const int* arow = adj + (size_t)i * N_NODES;
    for (int j = t; j < N_NODES; j += 256) {
        __builtin_prefetch(arow + j + 256, 0, 0);   // global_prefetch_b8
        int   av  = arow[j];
        float pen = (av == 0) ? NEG_INF : 0.0f;
#pragma unroll
        for (int h = 0; h < H_HEADS; ++h) {
            float e  = leaky(f1i[h] + f2[h * N_NODES + j]) + pen;
            float nm = fmaxf(m[h], e);
            s[h] = s[h] * __expf(m[h] - nm) + __expf(e - nm);
            m[h] = nm;
        }
    }
#pragma unroll
    for (int h = 0; h < H_HEADS; ++h) { shm[h][t] = m[h]; shs[h][t] = s[h]; }

    for (int off = 128; off > 0; off >>= 1) {
        __syncthreads();
        if (t < off) {
#pragma unroll
            for (int h = 0; h < H_HEADS; ++h) {
                float m1 = shm[h][t], m2 = shm[h][t + off];
                float M  = fmaxf(m1, m2);
                shs[h][t] = shs[h][t] * __expf(m1 - M) + shs[h][t + off] * __expf(m2 - M);
                shm[h][t] = M;
            }
        }
    }
    __syncthreads();
    if (t < H_HEADS) {
        rowM[t * N_NODES + i]  = shm[t][0];
        rowIS[t * N_NODES + i] = 1.0f / shs[t][0];
    }
}

// ---------------------------------------------------------------------------
// Kernel 4: flash pass 2 — out[h][i][:] = sum_j alpha[h][i][j] * h[h][j][:]
// All 8 waves of a block share one head, so the B tile (hT: 64 cols x 32 j,
// 4 KB) is staged once per block in LDS via double-buffered async
// global->LDS b128 copies (ASYNCcnt), then consumed with ds_load_b128.
// alpha tile is built in registers in exact WMMA A-layout.
// ---------------------------------------------------------------------------
#define BPAD 40   // padded halfs per column row in LDS (conflict-free b128 reads)

__global__ __launch_bounds__(256) void attn_gemm(const int* __restrict__ adj,
                                                 const float* __restrict__ f1,
                                                 const float* __restrict__ f2,
                                                 const float* __restrict__ rowM,
                                                 const float* __restrict__ rowIS,
                                                 const _Float16* __restrict__ hT,
                                                 float* __restrict__ out) {
    __shared__ _Float16 bsh[2][F_OUT * BPAD];

    int tid  = threadIdx.x;
    int wave = (blockIdx.x * blockDim.x + tid) >> 5;  // 1024 waves
    int lane = tid & 31;
    int head    = wave >> 8;      // all waves in a block share this
    int rowTile = wave & 255;
    int mrow = lane & 15;
    int kh   = lane >> 4;
    int i    = rowTile * 16 + mrow;

    float f1i = f1[head * N_NODES + i];
    float mi  = rowM[head * N_NODES + i];
    float is  = rowIS[head * N_NODES + i];

    const int*      arow = adj + (size_t)i * N_NODES;
    const float*    f2h  = f2 + head * N_NODES;
    const _Float16* hTh  = hT + (size_t)head * F_OUT * N_NODES;
    int ncol = lane & 15;

    // Per-thread staging slot: 256 threads x 16B = 4 KB tile per K-step.
    int scol   = tid >> 2;          // 0..63 column of hT
    int schunk = (tid & 3) * 8;     // 0,8,16,24 halfs within the 32-j run
    const _Float16* sg = hTh + (size_t)scol * N_NODES + schunk;
    _Float16*       sl = &bsh[0][scol * BPAD + schunk];

    v8f acc[4] = {v8f{}, v8f{}, v8f{}, v8f{}};

    // Prologue: stage tile 0 into buffer 0.
#if GAT_ASYNC
    __builtin_amdgcn_global_load_async_to_lds_b128((AS1 v4i*)(sg),
                                                   (AS3 v4i*)(sl), 0, 0);
#else
    *(v8h*)sl = *(const v8h*)sg;
#endif

    for (int t = 0; t < N_NODES / 32; ++t) {
        int jb = t * 32;
        if (t + 1 < N_NODES / 32) {
            _Float16* nl = sl + ((t + 1) & 1) * (F_OUT * BPAD);
#if GAT_ASYNC
            __builtin_amdgcn_global_load_async_to_lds_b128(
                (AS1 v4i*)(sg + (t + 1) * 32), (AS3 v4i*)(nl), 0, 0);
            __builtin_amdgcn_s_wait_asynccnt(1);   // tile t landed
#else
            *(v8h*)nl = *(const v8h*)(sg + (t + 1) * 32);
#endif
        } else {
#if GAT_ASYNC
            __builtin_amdgcn_s_wait_asynccnt(0);   // last tile landed
#endif
        }
        __syncthreads();   // tile t visible to all waves

        // ---- A: alpha 16x32 tile in registers (WMMA A-layout).
        int j1 = jb + 8 * kh;        // elems 0..7  (K = 8kh..8kh+7)
        int j2 = jb + 16 + 8 * kh;   // elems 8..15 (K = 16+8kh..+7)
        int av[16];
        *(int4*)(av + 0)  = *(const int4*)(arow + j1);
        *(int4*)(av + 4)  = *(const int4*)(arow + j1 + 4);
        *(int4*)(av + 8)  = *(const int4*)(arow + j2);
        *(int4*)(av + 12) = *(const int4*)(arow + j2 + 4);
        float fv[16];
        *(float4*)(fv + 0)  = *(const float4*)(f2h + j1);
        *(float4*)(fv + 4)  = *(const float4*)(f2h + j1 + 4);
        *(float4*)(fv + 8)  = *(const float4*)(f2h + j2);
        *(float4*)(fv + 12) = *(const float4*)(f2h + j2 + 4);

        v16h A;
#pragma unroll
        for (int e = 0; e < 16; ++e) {
            float ev = leaky(f1i + fv[e]);
            float al = (av[e] == 0) ? 0.0f : __expf(ev - mi) * is;
            A[e] = (_Float16)al;
        }

        // ---- B from LDS: lane reads K = 8kh..+7 and 16+8kh..+7 halfs.
        const _Float16* bbuf = &bsh[t & 1][0];
#pragma unroll
        for (int tcol = 0; tcol < 4; ++tcol) {
            const _Float16* bp = bbuf + (tcol * 16 + ncol) * BPAD + 8 * kh;
            v8h b0 = *(const v8h*)(bp);
            v8h b1 = *(const v8h*)(bp + 16);
            v16h B;
#pragma unroll
            for (int e = 0; e < 8; ++e) { B[e] = b0[e]; B[8 + e] = b1[e]; }
            acc[tcol] = __builtin_amdgcn_wmma_f32_16x16x32_f16(
                false, A, false, B, (short)0, acc[tcol], false, false);
        }

        __syncthreads();   // everyone done reading buf (t&1) before overwrite
    }

    // Store: out[n][head*64 + tcol*16 + ncol], n = rowTile*16 + r + 8*kh
#pragma unroll
    for (int tcol = 0; tcol < 4; ++tcol) {
#pragma unroll
        for (int r = 0; r < 8; ++r) {
            int n = rowTile * 16 + r + 8 * kh;
            out[(size_t)n * (H_HEADS * F_OUT) + head * F_OUT + tcol * 16 + ncol] =
                acc[tcol][r];
        }
    }
}

// ---------------------------------------------------------------------------
extern "C" void kernel_launch(void* const* d_in, const int* in_sizes, int n_in,
                              void* d_out, int out_size, void* d_ws, size_t ws_size,
                              hipStream_t stream) {
    const float* x   = (const float*)d_in[0];
    const int*   adj = (const int*)d_in[1];
    const float* W   = (const float*)d_in[2];
    const float* a   = (const float*)d_in[3];
    float*       out = (float*)d_out;

    char* ws = (char*)d_ws;
    size_t off = 0;
    _Float16* hT = (_Float16*)(ws + off);  off += (size_t)H_HEADS * F_OUT * N_NODES * sizeof(_Float16); // 2 MB
    float* f1    = (float*)(ws + off);     off += (size_t)H_HEADS * N_NODES * sizeof(float);
    float* f2    = (float*)(ws + off);     off += (size_t)H_HEADS * N_NODES * sizeof(float);
    float* rowM  = (float*)(ws + off);     off += (size_t)H_HEADS * N_NODES * sizeof(float);
    float* rowIS = (float*)(ws + off);     off += (size_t)H_HEADS * N_NODES * sizeof(float);
    _Float16* WT = (_Float16*)(ws + off);  off += (size_t)H_HEADS * F_OUT * F_IN * sizeof(_Float16);

    prep_wt  <<<(H_HEADS * F_IN * F_OUT) / 256, 256, 0, stream>>>(W, WT);
    gemm_h   <<<512, 256, 0, stream>>>(x, WT, hT);               // 4096 waves
    f1f2_k   <<<(H_HEADS * N_NODES) / 256, 256, 0, stream>>>(hT, a, f1, f2);
    row_stats<<<N_NODES, 256, 0, stream>>>(adj, f1, f2, rowM, rowIS);
    attn_gemm<<<128, 256, 0, stream>>>(adj, f1, f2, rowM, rowIS, hT, out); // 1024 waves
}
